// EEG_GAT_77610059038988
// MI455X (gfx1250) — matrix operand order repaired and checked by reading, not compiled, not run on hardware
//
#include <hip/hip_runtime.h>

// GAT fused kernel for MI455X (gfx1250, wave32, WMMA bf16).
// One workgroup (256 threads = 8 waves) handles one batch element:
//   h = x*W  (64x256 = 64x256 * 256x256, bf16 WMMA, f32 accum)
//   scores + softmax (f32, LDS)
//   out = alpha*h + bias (64x256 = 64x64 * 64x256, bf16 WMMA)

#define C_NODES 63
#define F_INN   250
#define F_OUTT  250
#define NEG_SLOPE 0.2f

typedef __attribute__((ext_vector_type(16))) __bf16 v16bf;
typedef __attribute__((ext_vector_type(8)))  float  v8f;

union FragB16 { v16bf v; unsigned u[8]; };

// native hardware bf16 conversion (v_cvt_pk_bf16_f32 class ops)
static __device__ __forceinline__ unsigned short f2bf(float f) {
    union { __bf16 h; unsigned short s; } c; c.h = (__bf16)f; return c.s;
}
static __device__ __forceinline__ unsigned pack2bf(float lo, float hi) {
    union { __bf16 h[2]; unsigned u; } p;
    p.h[0] = (__bf16)lo; p.h[1] = (__bf16)hi; return p.u;
}
static __device__ __forceinline__ float b2f(unsigned short s) {
    union { unsigned short s; __bf16 h; } c; c.s = s; return (float)c.h;
}
static __device__ __forceinline__ v8f v8f_zero() {
    v8f z = {0.f,0.f,0.f,0.f,0.f,0.f,0.f,0.f}; return z;
}

// LDS layout (bytes):
//  w_t   : 258*256 u16 = 132096   (W transposed: w_t[n*258 + k], bf16, zero padded)
//  x_lds : 258*64  u16 =  33024   (x row-major: x[r*258 + k], bf16, zero padded)
//  ht    : 66*256  u16 =  33792   (h col-major: ht[c*66 + r], bf16)
//  alpha : 66*64   u16 =   8448   (alpha row-major: alpha[d*66 + s], bf16)
//  attS/attD/bias : 3*256 f32 = 3072
//  aS/aD : 2*64 f32 = 512
//  pS/pD : 2*256 f32 = 2048
#define OFF_W     0
#define OFF_X     132096
#define OFF_HT    (132096 + 33024)
#define OFF_ALPHA (OFF_HT + 33792)
#define OFF_ATTS  (OFF_ALPHA + 8448)
#define SMEM_BYTES 212992

__global__ __launch_bounds__(256, 1)
void gat_fused_kernel(const float* __restrict__ xg,
                      const float* __restrict__ Wg,
                      const float* __restrict__ attSg,
                      const float* __restrict__ attDg,
                      const float* __restrict__ biasg,
                      float* __restrict__ outg)
{
    extern __shared__ char smem[];
    unsigned short* w_t  = (unsigned short*)(smem + OFF_W);
    unsigned short* xls  = (unsigned short*)(smem + OFF_X);
    unsigned short* ht   = (unsigned short*)(smem + OFF_HT);
    unsigned short* alp  = (unsigned short*)(smem + OFF_ALPHA);
    float* attS = (float*)(smem + OFF_ATTS);
    float* attD = attS + 256;
    float* biasL = attD + 256;
    float* aS = biasL + 256;
    float* aD = aS + 64;
    float* pS = aD + 64;    // [64][4]
    float* pD = pS + 256;   // [64][4]

    const int tid  = threadIdx.x;
    const int b    = blockIdx.x;
    const int lane = tid & 31;
    const int wv   = tid >> 5;          // wave 0..7
    const int mt   = wv & 3;            // M tile (rows mt*16..mt*16+15)
    const int nbase = (wv >> 2) * 8;    // N tiles nbase..nbase+7
    const int hi   = lane >> 4;         // half-wave select
    const int lrow = lane & 15;

    // ---- prefetch W into cache hierarchy while we zero LDS ----
    for (int off = tid * 64; off < F_INN * F_OUTT; off += 256 * 64)
        __builtin_prefetch(Wg + off, 0, 0);

    // ---- zero padded staging regions (w_t + x_lds) ----
    {
        unsigned* z = (unsigned*)smem;
        const int nzero = (OFF_HT) / 4;
        for (int i = tid; i < nzero; i += 256) z[i] = 0u;
        for (int i = tid; i < 256; i += 256) {
            attS[i]  = (i < F_OUTT) ? attSg[i]  : 0.f;
            attD[i]  = (i < F_OUTT) ? attDg[i]  : 0.f;
            biasL[i] = (i < F_OUTT) ? biasg[i]  : 0.f;
        }
    }
    __syncthreads();

    // ---- stage W transposed as bf16 (coalesced global reads) ----
    {
        int k = tid / F_OUTT;
        int n = tid - k * F_OUTT;
        for (int idx = tid; idx < F_INN * F_OUTT; idx += 256) {
            w_t[n * 258 + k] = f2bf(Wg[idx]);
            n += 6; ++k;                       // advance by 256 = 250 + 6
            if (n >= F_OUTT) { n -= F_OUTT; ++k; }
        }
    }
    // ---- stage x[b] as bf16 row-major ----
    {
        const float* xb = xg + (size_t)b * (C_NODES * F_INN);
        int r = tid / F_INN;
        int c = tid - r * F_INN;
        for (int idx = tid; idx < C_NODES * F_INN; idx += 256) {
            xls[r * 258 + c] = f2bf(xb[idx]);
            c += 6; ++r;                       // advance by 256 = 250 + 6
            if (c >= F_INN) { c -= F_INN; ++r; }
        }
    }
    __syncthreads();

    // ================= GEMM1: h = x * W =================
    const unsigned* xw = (const unsigned*)xls;   // row stride 129 dwords
    const unsigned* ww = (const unsigned*)w_t;   // col stride 129 dwords
    const int row = mt * 16 + lrow;

    v8f acc[8];
#pragma unroll
    for (int n = 0; n < 8; ++n) acc[n] = v8f_zero();

#pragma unroll 1
    for (int kt = 0; kt < 8; ++kt) {
        FragB16 a;
        int abase = row * 129 + kt * 16 + hi * 4;
#pragma unroll
        for (int i = 0; i < 4; ++i) { a.u[i] = xw[abase + i]; a.u[4 + i] = xw[abase + 8 + i]; }
#pragma unroll
        for (int n = 0; n < 8; ++n) {
            int col = (nbase + n) * 16 + lrow;
            int bbase = col * 129 + kt * 16 + hi * 4;
            FragB16 bb;
#pragma unroll
            for (int i = 0; i < 4; ++i) { bb.u[i] = ww[bbase + i]; bb.u[4 + i] = ww[bbase + 8 + i]; }
            acc[n] = __builtin_amdgcn_wmma_f32_16x16x32_bf16(
                false, a.v, false, bb.v, (short)0, acc[n], false, false);
        }
    }

    // ---- write h to LDS column-major bf16: ht[c*66 + r] ----
    {
        unsigned* htw = (unsigned*)ht;           // col stride 33 dwords
        int r0 = mt * 16 + hi * 8;
#pragma unroll
        for (int n = 0; n < 8; ++n) {
            int col = (nbase + n) * 16 + lrow;
            int base = col * 33 + (r0 >> 1);
#pragma unroll
            for (int i = 0; i < 4; ++i)
                htw[base + i] = pack2bf(acc[n][2 * i], acc[n][2 * i + 1]);
        }
    }
    __syncthreads();

    // ================= attention scores =================
    {
        int r = tid & 63;
        int q = tid >> 6;
        float ss = 0.f, sd = 0.f;
        int c0 = q * 64;
        for (int c = c0; c < c0 + 64; ++c) {
            float hv = b2f(ht[c * 66 + r]);
            ss += hv * attS[c];
            sd += hv * attD[c];
        }
        pS[r * 4 + q] = ss;
        pD[r * 4 + q] = sd;
    }
    __syncthreads();
    if (tid < 64) {
        aS[tid] = pS[tid * 4] + pS[tid * 4 + 1] + pS[tid * 4 + 2] + pS[tid * 4 + 3];
        aD[tid] = pD[tid * 4] + pD[tid * 4 + 1] + pD[tid * 4 + 2] + pD[tid * 4 + 3];
    }
    __syncthreads();

    // ================= softmax -> alpha (bf16) =================
    if (tid < 64) {
        if (tid < C_NODES) {
            float ad = aD[tid];
            float m = -1e30f;
            for (int s = 0; s < C_NODES; ++s) {
                float v = aS[s] + ad;
                v = v > 0.f ? v : NEG_SLOPE * v;
                m = fmaxf(m, v);
            }
            float sum = 0.f;
            for (int s = 0; s < C_NODES; ++s) {
                float v = aS[s] + ad;
                v = v > 0.f ? v : NEG_SLOPE * v;
                sum += __expf(v - m);
            }
            float inv = 1.f / sum;
            for (int s = 0; s < C_NODES; ++s) {
                float v = aS[s] + ad;
                v = v > 0.f ? v : NEG_SLOPE * v;
                alp[tid * 66 + s] = f2bf(__expf(v - m) * inv);
            }
            alp[tid * 66 + 63] = 0;  // padded source column
        } else {
            for (int s = 0; s < 64; ++s) alp[63 * 66 + s] = 0;  // padded dst row
        }
    }
    __syncthreads();

    // ================= GEMM2: out = alpha * h =================
    const unsigned* aw = (const unsigned*)alp;   // row stride 33 dwords
    const unsigned* hw = (const unsigned*)ht;    // col stride 33 dwords

    v8f acc2[8];
#pragma unroll
    for (int n = 0; n < 8; ++n) acc2[n] = v8f_zero();

#pragma unroll
    for (int kt = 0; kt < 2; ++kt) {
        FragB16 a;
        int abase = row * 33 + kt * 16 + hi * 4;
#pragma unroll
        for (int i = 0; i < 4; ++i) { a.u[i] = aw[abase + i]; a.u[4 + i] = aw[abase + 8 + i]; }
#pragma unroll
        for (int n = 0; n < 8; ++n) {
            int col = (nbase + n) * 16 + lrow;
            int bbase = col * 33 + kt * 16 + hi * 4;
            FragB16 bb;
#pragma unroll
            for (int i = 0; i < 4; ++i) { bb.u[i] = hw[bbase + i]; bb.u[4 + i] = hw[bbase + 8 + i]; }
            acc2[n] = __builtin_amdgcn_wmma_f32_16x16x32_bf16(
                false, a.v, false, bb.v, (short)0, acc2[n], false, false);
        }
    }

    // ---- add bias, store f32 output (guarded for padding) ----
    {
        float* outb = outg + (size_t)b * (C_NODES * F_OUTT);
        int r0 = mt * 16 + hi * 8;
#pragma unroll
        for (int n = 0; n < 8; ++n) {
            int col = (nbase + n) * 16 + lrow;
            if (col < F_OUTT) {
                float bv = biasL[col];
#pragma unroll
                for (int r = 0; r < 8; ++r) {
                    int rr = r0 + r;
                    if (rr < C_NODES) outb[rr * F_OUTT + col] = acc2[n][r] + bv;
                }
            }
        }
    }
}

extern "C" void kernel_launch(void* const* d_in, const int* in_sizes, int n_in,
                              void* d_out, int out_size, void* d_ws, size_t ws_size,
                              hipStream_t stream) {
    const float* x    = (const float*)d_in[0];
    const float* W    = (const float*)d_in[1];
    const float* atts = (const float*)d_in[2];
    const float* attd = (const float*)d_in[3];
    const float* bias = (const float*)d_in[4];
    float* out = (float*)d_out;

    int batches = in_sizes[0] / (C_NODES * F_INN);   // 4096
    dim3 grid(batches), block(256);
    gat_fused_kernel<<<grid, block, SMEM_BYTES, stream>>>(x, W, atts, attd, bias, out);
}
